// HungarianMatcher_40587440947913
// MI455X (gfx1250) — matrix operand order repaired and checked by reading, not compiled, not run on hardware
//
#include <hip/hip_runtime.h>
#include <hip/hip_bf16.h>

// ---------------------------------------------------------------------------
// Problem constants (from the reference)
// ---------------------------------------------------------------------------
#define BB     4
#define QQ     100
#define QP     112      // Q padded to multiple of 16
#define CC     81
#define TT     50
#define TP     64       // T padded to multiple of 16
#define HWN    65536    // H*W
#define KSPLIT 64       // K-slices per GEMM row-block (each slice = 1024)
#define KSLICE (HWN / KSPLIT)

typedef __attribute__((ext_vector_type(16))) _Float16 v16h;
typedef __attribute__((ext_vector_type(8)))  float    v8f;

union Frag16 { uint4 u[2]; v16h h; };

// ---------------------------------------------------------------------------
// Kernel 1: per-(b,q) feature prep for masks & iams.
//   Am = focal_pos - focal_neg (masks), Ai = same (iams), As = sigmoid(masks)
//   rs_m = sum focal_neg(mask), rs_i = sum focal_neg(iam), rs_s = sum sigmoid
// One 1024-thread block per (b,q) row -> deterministic LDS reduction.
// ---------------------------------------------------------------------------
__global__ __launch_bounds__(1024)
void k_prepare(const float* __restrict__ pm, const float* __restrict__ pi,
               _Float16* __restrict__ Am, _Float16* __restrict__ Ai,
               _Float16* __restrict__ As,
               float* __restrict__ rs_m, float* __restrict__ rs_i,
               float* __restrict__ rs_s)
{
    const int row = blockIdx.x;               // b*QQ + q
    const int b   = row / QQ;
    const int q   = row % QQ;
    const size_t inBase  = (size_t)row * HWN;
    const size_t outBase = ((size_t)b * QP + q) * HWN;
    const int tid = threadIdx.x;

    float sm = 0.f, si = 0.f, ss = 0.f;
    for (int k = tid; k < HWN; k += 1024) {
        // masks
        float x   = pm[inBase + k];
        float p   = 1.f / (1.f + __expf(-x));
        float spp = fmaxf(x, 0.f) + __logf(1.f + __expf(-fabsf(x))); // softplus(x)
        float spn = spp - x;                                         // softplus(-x)
        float fp  = 0.25f * (1.f - p) * (1.f - p) * spn;
        float fn  = 0.75f * p * p * spp;
        Am[outBase + k] = (_Float16)(fp - fn);
        As[outBase + k] = (_Float16)p;
        sm += fn; ss += p;
        // iams
        float y    = pi[inBase + k];
        float py   = 1.f / (1.f + __expf(-y));
        float sppy = fmaxf(y, 0.f) + __logf(1.f + __expf(-fabsf(y)));
        float spny = sppy - y;
        float fpi  = 0.25f * (1.f - py) * (1.f - py) * spny;
        float fni  = 0.75f * py * py * sppy;
        Ai[outBase + k] = (_Float16)(fpi - fni);
        si += fni;
    }

    __shared__ float sh0[1024], sh1[1024], sh2[1024];
    sh0[tid] = sm; sh1[tid] = si; sh2[tid] = ss;
    __syncthreads();
    for (int s = 512; s > 0; s >>= 1) {
        if (tid < s) { sh0[tid] += sh0[tid + s]; sh1[tid] += sh1[tid + s]; sh2[tid] += sh2[tid + s]; }
        __syncthreads();
    }
    if (tid == 0) { rs_m[row] = sh0[0]; rs_i[row] = sh1[0]; rs_s[row] = sh2[0]; }
}

// ---------------------------------------------------------------------------
// Kernel 2: target masks -> f16 B plane [b][t_pad][HW] (N-major) + column sums
// ---------------------------------------------------------------------------
__global__ __launch_bounds__(256)
void k_tgt(const int* __restrict__ tm, _Float16* __restrict__ Bt,
           float* __restrict__ tsum)
{
    const int row = blockIdx.x;            // b*TP + tp
    const int b   = row >> 6;
    const int tp  = row & 63;
    const int tid = threadIdx.x;
    const size_t outBase = (size_t)row * HWN;

    float s = 0.f;
    if (tp < TT) {
        const size_t inBase = ((size_t)b * TT + tp) * HWN;
        for (int k = tid; k < HWN; k += 256) {
            int v = tm[inBase + k];
            Bt[outBase + k] = (_Float16)v;
            s += (float)v;
        }
    } else {
        for (int k = tid; k < HWN; k += 256) Bt[outBase + k] = (_Float16)0;
    }

    __shared__ float sh[256];
    sh[tid] = s;
    __syncthreads();
    for (int st = 128; st > 0; st >>= 1) {
        if (tid < st) sh[tid] += sh[tid + st];
        __syncthreads();
    }
    if (tid == 0) tsum[row] = sh[0];
}

// ---------------------------------------------------------------------------
// Kernel 3: zero the Q-padding rows (q in [100,112)) of the three A planes
// ---------------------------------------------------------------------------
__global__ __launch_bounds__(256)
void k_padzero(_Float16* __restrict__ Am, _Float16* __restrict__ Ai,
               _Float16* __restrict__ As)
{
    const int a   = blockIdx.y;
    _Float16* A   = (a == 0) ? Am : ((a == 1) ? Ai : As);
    const int idx = blockIdx.x * blockDim.x + threadIdx.x; // uint4 index in pad region
    const int perB = (QP - QQ) * (HWN / 8);                // 12 * 8192
    if (idx >= BB * perB) return;
    const int b   = idx / perB;
    const int rem = idx % perB;
    uint4* p = (uint4*)A;
    p[(size_t)(b * QP + QQ) * (HWN / 8) + rem] = make_uint4(0u, 0u, 0u, 0u);
}

// ---------------------------------------------------------------------------
// Kernel 4: WMMA GEMM, software-pipelined (double-buffered fragments).
// One wave per (b, qtile, gemm, kslice):
//   D(16x64) += A(16x1024 slice, f16) * B(1024x64 slice, f16), f32 accum.
// A fragment layout (16-bit A 16x32): lane r=L&15 holds row M=r; hi=L>>4 picks
// K = hi*16..hi*16+15 (contiguous, matching B) -> two 16B loads.
// B fragment layout: lane n holds column N=n&15; lanes 0-15: K=0..15,
// lanes 16-31: K=16..31 -> one contiguous 32B (two 16B loads) from N-major Bt.
// The ping-pong structure lets the next K-step's 10 b128 loads fly while the
// current 4 WMMAs execute (partial s_wait_loadcnt instead of full drains).
// ---------------------------------------------------------------------------
__global__ __launch_bounds__(32)
void k_gemm(const _Float16* __restrict__ Am, const _Float16* __restrict__ Ai,
            const _Float16* __restrict__ As, const _Float16* __restrict__ Bt,
            float* __restrict__ Gs)
{
    const int lane = threadIdx.x;
    const int bq   = blockIdx.x;          // b*7 + qtile
    const int b    = bq / 7;
    const int qt   = bq % 7;
    const int g    = blockIdx.y;          // which GEMM (0:mask,1:iam,2:sigmoid)
    const int ks   = blockIdx.z;          // K slice

    const _Float16* A = (g == 0) ? Am : ((g == 1) ? Ai : As);
    const int r  = lane & 15;
    const int hi = lane >> 4;

    const _Float16* Arow = A  + ((size_t)(b * QP + qt * 16 + r)) * HWN;
    const _Float16* Bcol = Bt + ((size_t)(b * TP) + (lane & 15)) * HWN;
    const int k0base = ks * KSLICE;

    v8f acc0 = {}, acc1 = {}, acc2 = {}, acc3 = {};

    Frag16 a0, b00, b01, b02, b03;   // buffer 0
    Frag16 a1, b10, b11, b12, b13;   // buffer 1

    auto LD = [&](Frag16& a, Frag16& c0, Frag16& c1, Frag16& c2, Frag16& c3,
                  int k0) {
        a.u[0] = *(const uint4*)(Arow + k0 + hi * 16);
        a.u[1] = *(const uint4*)(Arow + k0 + hi * 16 + 8);
        const _Float16* bp = Bcol + k0 + hi * 16;
        c0.u[0] = *(const uint4*)(bp);
        c0.u[1] = *(const uint4*)(bp + 8);
        c1.u[0] = *(const uint4*)(bp + (size_t)16 * HWN);
        c1.u[1] = *(const uint4*)(bp + (size_t)16 * HWN + 8);
        c2.u[0] = *(const uint4*)(bp + (size_t)32 * HWN);
        c2.u[1] = *(const uint4*)(bp + (size_t)32 * HWN + 8);
        c3.u[0] = *(const uint4*)(bp + (size_t)48 * HWN);
        c3.u[1] = *(const uint4*)(bp + (size_t)48 * HWN + 8);
    };
    auto MMA = [&](const Frag16& a, const Frag16& c0, const Frag16& c1,
                   const Frag16& c2, const Frag16& c3) {
        acc0 = __builtin_amdgcn_wmma_f32_16x16x32_f16(false, a.h, false, c0.h,
                                                      (short)0, acc0, false, false);
        acc1 = __builtin_amdgcn_wmma_f32_16x16x32_f16(false, a.h, false, c1.h,
                                                      (short)0, acc1, false, false);
        acc2 = __builtin_amdgcn_wmma_f32_16x16x32_f16(false, a.h, false, c2.h,
                                                      (short)0, acc2, false, false);
        acc3 = __builtin_amdgcn_wmma_f32_16x16x32_f16(false, a.h, false, c3.h,
                                                      (short)0, acc3, false, false);
    };

    // prologue: fill buffer 0
    LD(a0, b00, b01, b02, b03, k0base);
    int kk = 0;
    // steady state: ping-pong, loads for step i+1 issued before MMAs of step i
    for (; kk < KSLICE - 64; kk += 64) {
        LD(a1, b10, b11, b12, b13, k0base + kk + 32);
        MMA(a0, b00, b01, b02, b03);
        LD(a0, b00, b01, b02, b03, k0base + kk + 64);
        MMA(a1, b10, b11, b12, b13);
    }
    // epilogue: last pair (kk == KSLICE - 64)
    LD(a1, b10, b11, b12, b13, k0base + kk + 32);
    MMA(a0, b00, b01, b02, b03);
    MMA(a1, b10, b11, b12, b13);

    // C/D layout: VGPR i, lane L -> M = i + (L>=16)*8, N = L&15
    float* Go = Gs + (((size_t)(b * 3 + g) * KSPLIT + ks) * QP * TP);
    const int n = lane & 15;
#pragma unroll
    for (int i = 0; i < 8; i++) {
        const int m = qt * 16 + i + hi * 8;
        Go[(size_t)m * TP + n +  0] = acc0[i];
        Go[(size_t)m * TP + n + 16] = acc1[i];
        Go[(size_t)m * TP + n + 32] = acc2[i];
        Go[(size_t)m * TP + n + 48] = acc3[i];
    }
}

// ---------------------------------------------------------------------------
// Kernel 5: epilogue. Sum K-slice partials, fuse class cost / dice / weights.
// ---------------------------------------------------------------------------
__global__ __launch_bounds__(256)
void k_epilogue(const float* __restrict__ logits, const int* __restrict__ tgt_ids,
                const float* __restrict__ Gs,
                const float* __restrict__ rs_m, const float* __restrict__ rs_i,
                const float* __restrict__ rs_s, const float* __restrict__ tsum,
                float* __restrict__ out)
{
    const int idx = blockIdx.x * blockDim.x + threadIdx.x;
    if (idx >= BB * QQ * TT) return;
    const int t = idx % TT;
    const int q = (idx / TT) % QQ;
    const int b = idx / (QQ * TT);

    // --- class cost (softmax over C=81, gathered at target label) ---
    const float* lg = logits + ((size_t)b * QQ + q) * CC;
    float mx = -3.4e38f;
    for (int c = 0; c < CC; c++) mx = fmaxf(mx, lg[c]);
    float den = 0.f;
    for (int c = 0; c < CC; c++) den += __expf(lg[c] - mx);
    const int cls = tgt_ids[b * TT + t];
    const float p = __expf(lg[cls] - mx) / den;
    const float pos = 0.25f * (1.f - p) * (1.f - p) * (-__logf(p + 1e-8f));
    const float neg = 0.75f * p * p * (-__logf(1.f - p + 1e-8f));
    const float cost_class = pos - neg;

    // --- reduce K-slice partials for the three GEMMs ---
    const size_t stride = (size_t)QP * TP;
    const size_t base   = (size_t)q * TP + t;
    const float* G0 = Gs + ((size_t)(b * 3 + 0) * KSPLIT) * stride + base;
    const float* G1 = Gs + ((size_t)(b * 3 + 1) * KSPLIT) * stride + base;
    const float* G2 = Gs + ((size_t)(b * 3 + 2) * KSPLIT) * stride + base;
    float gm = 0.f, gi = 0.f, gd = 0.f;
    for (int s = 0; s < KSPLIT; s++) {
        gm += G0[(size_t)s * stride];
        gi += G1[(size_t)s * stride];
        gd += G2[(size_t)s * stride];
    }

    const float inv = 1.f / (float)HWN;
    const float cost_mask = (gm + rs_m[b * QQ + q]) * inv;
    const float cost_iams = (gi + rs_i[b * QQ + q]) * inv;
    const float cost_dice = 1.f - (2.f * gd + 1.f) /
                                  (rs_s[b * QQ + q] + tsum[b * TP + t] + 1.f);

    out[idx] = 5.f * cost_mask + 5.f * cost_iams + 2.f * cost_class + 2.f * cost_dice;
}

// ---------------------------------------------------------------------------
// Host-side launcher
// ---------------------------------------------------------------------------
extern "C" void kernel_launch(void* const* d_in, const int* in_sizes, int n_in,
                              void* d_out, int out_size, void* d_ws, size_t ws_size,
                              hipStream_t stream)
{
    const float* pred_logits = (const float*)d_in[0];
    const float* pred_masks  = (const float*)d_in[1];
    const float* pred_iams   = (const float*)d_in[2];
    const int*   tgt_ids     = (const int*)d_in[3];
    const int*   tgt_masks   = (const int*)d_in[4];
    float*       out         = (float*)d_out;

    // workspace layout
    char* ws = (char*)d_ws;
    const size_t szA  = (size_t)BB * QP * HWN * sizeof(_Float16); // 58.7 MB each
    const size_t szBt = (size_t)BB * TP * HWN * sizeof(_Float16); // 33.6 MB
    const size_t szGs = (size_t)BB * 3 * KSPLIT * QP * TP * sizeof(float); // 22 MB

    _Float16* Am  = (_Float16*)(ws);
    _Float16* Ai  = (_Float16*)(ws + szA);
    _Float16* As  = (_Float16*)(ws + 2 * szA);
    _Float16* Bt  = (_Float16*)(ws + 3 * szA);
    float*    Gs  = (float*)   (ws + 3 * szA + szBt);
    float*    rs_m = (float*)  (ws + 3 * szA + szBt + szGs);
    float*    rs_i = rs_m + BB * QQ;
    float*    rs_s = rs_i + BB * QQ;
    float*    tsum = rs_s + BB * QQ;

    // 1) feature prep: one block per (b,q) row
    k_prepare<<<dim3(BB * QQ), dim3(1024), 0, stream>>>(
        pred_masks, pred_iams, Am, Ai, As, rs_m, rs_i, rs_s);

    // 2) target prep: one block per (b, t_pad) row
    k_tgt<<<dim3(BB * TP), dim3(256), 0, stream>>>(tgt_masks, Bt, tsum);

    // 3) zero Q-pad rows of the A planes
    const int padVec4 = BB * (QP - QQ) * (HWN / 8); // 393216
    k_padzero<<<dim3((padVec4 + 255) / 256, 3), dim3(256), 0, stream>>>(Am, Ai, As);

    // 4) WMMA GEMMs: grid = (b*qtiles, gemm, kslice), 1 wave each
    k_gemm<<<dim3(BB * (QP / 16), 3, KSPLIT), dim3(32), 0, stream>>>(
        Am, Ai, As, Bt, Gs);

    // 5) epilogue
    const int nOut = BB * QQ * TT;
    k_epilogue<<<dim3((nOut + 255) / 256), dim3(256), 0, stream>>>(
        pred_logits, tgt_ids, Gs, rs_m, rs_i, rs_s, tsum, out);
}
